// SAE_LTM_69982197121802
// MI455X (gfx1250) — matrix operand order repaired and checked by reading, not compiled, not run on hardware
//
#include <hip/hip_runtime.h>
#include <hip/hip_bf16.h>
#include <math.h>
#include <stdint.h>

// Problem constants (from reference)
#define B_   8192
#define SD_  1024
#define IN_  512
#define HID_ 4096

typedef __attribute__((ext_vector_type(16))) __bf16 v16bf;
typedef __attribute__((ext_vector_type(8)))  float  v8f;

#define TILE_M 128
#define TILE_N 128
#define TILE_K 64
#define ROW_B   144                // LDS row stride bytes (64 bf16 data + pad): 16B aligned, conflict-free
#define ROW_E   (ROW_B / 2)        // in bf16 elements
#define LDSA_BYTES (TILE_M * ROW_B)   // 18432
#define LDSB_BYTES (TILE_N * ROW_B)   // 18432

__device__ __forceinline__ unsigned lds_off(const void* p) {
  // generic pointer to LDS: low 32 bits are the workgroup-relative LDS offset
  return (unsigned)(uintptr_t)p;
}

__device__ __forceinline__ void async_copy_b128(unsigned lds_byte_off, const void* gaddr) {
  asm volatile("global_load_async_to_lds_b128 %0, %1, off"
               :: "v"(lds_byte_off), "v"((unsigned long long)(uintptr_t)gaddr)
               : "memory");
}

// ---------------------------------------------------------------------------
// GEMM: C[M,N] = act(A[M,K] @ Bt[N,K]^T + bias[N])
// A, Bt bf16 K-major; async-to-LDS double-buffered staging; bf16 WMMA f32 acc.
// Block tile 128x128xK64; 8 waves as 4(M) x 2(N); wave tile 32x64 = 2x4 WMMA.
// ACT: 0 = identity, 1 = SiLU, 2 = ReLU.  All dims divide tiles exactly here.
// ---------------------------------------------------------------------------
template<int ACT, bool OUT_BF16>
__global__ __launch_bounds__(256, 2)
void gemm_async(const __bf16* __restrict__ A,   // [M,K]
                const __bf16* __restrict__ Bt,  // [N,K]
                const float*  __restrict__ bias,
                void* __restrict__ Cout,
                int M, int N, int K)
{
  __shared__ __bf16 lA[2][TILE_M * ROW_E];
  __shared__ __bf16 lB[2][TILE_N * ROW_E];

  const int tid   = threadIdx.x;
  const int lane  = tid & 31;
  const int wave  = tid >> 5;       // 0..7
  const int waveM = wave & 3;       // 0..3 -> 32-row strip
  const int waveN = wave >> 2;      // 0..1 -> 64-col strip
  const int half  = lane >> 4;
  const int l16   = lane & 15;

  const int m0 = blockIdx.y * TILE_M;
  const int n0 = blockIdx.x * TILE_N;

  const unsigned ldsA0 = lds_off(&lA[0][0]);
  const unsigned ldsB0 = lds_off(&lB[0][0]);

  // stage one 128x64 + 128x64 K-tile pair into buffer `buf` (8 asyncs/wave)
  auto stage = [&](int buf, int k0) {
#pragma unroll
    for (int i = 0; i < 4; ++i) {
      const int c   = tid + i * 256;       // 1024 x 16B chunks for A
      const int row = c >> 3, ch = c & 7;
      async_copy_b128(ldsA0 + (unsigned)buf * LDSA_BYTES + row * ROW_B + ch * 16,
                      (const char*)(A + (size_t)(m0 + row) * K + k0) + ch * 16);
    }
#pragma unroll
    for (int i = 0; i < 4; ++i) {
      const int c   = tid + i * 256;       // 1024 x 16B chunks for B
      const int row = c >> 3, ch = c & 7;
      async_copy_b128(ldsB0 + (unsigned)buf * LDSB_BYTES + row * ROW_B + ch * 16,
                      (const char*)(Bt + (size_t)(n0 + row) * K + k0) + ch * 16);
    }
  };

  v8f acc[2][4] = {};

  // 16 WMMAs on buffer `buf` (2 K-sub-steps of 32)
  auto compute = [&](int buf) {
    const __bf16* sA = &lA[buf][0];
    const __bf16* sB = &lB[buf][0];
#pragma unroll
    for (int ks = 0; ks < 2; ++ks) {
      v16bf af[2], bfm[4];
      // A fragment (16x32): v<4 -> K=half*8+2v(+1); v>=4 -> K=16+half*8+2(v-4)(+1)
#pragma unroll
      for (int i = 0; i < 2; ++i) {
        const __bf16* pr = sA + (size_t)(waveM * 32 + i * 16 + l16) * ROW_E + ks * 32;
#pragma unroll
        for (int v = 0; v < 8; ++v) {
          const int kk = ((v < 4) ? 0 : 16) + half * 8 + (v & 3) * 2;
          af[i][2 * v + 0] = pr[kk + 0];
          af[i][2 * v + 1] = pr[kk + 1];
        }
      }
      // B fragment (32x16): lane -> N=l16; v<4 -> K=half*8+2v(+1); v>=4 -> K=16+half*8+2(v-4)(+1)
#pragma unroll
      for (int j = 0; j < 4; ++j) {
        const __bf16* pr = sB + (size_t)(waveN * 64 + j * 16 + l16) * ROW_E + ks * 32;
#pragma unroll
        for (int v = 0; v < 8; ++v) {
          const int kk = ((v < 4) ? 0 : 16) + half * 8 + (v & 3) * 2;
          bfm[j][2 * v + 0] = pr[kk + 0];
          bfm[j][2 * v + 1] = pr[kk + 1];
        }
      }
#pragma unroll
      for (int i = 0; i < 2; ++i)
#pragma unroll
        for (int j = 0; j < 4; ++j)
          acc[i][j] = __builtin_amdgcn_wmma_f32_16x16x32_bf16(
              false, af[i], false, bfm[j], (short)0, acc[i][j], false, false);
    }
  };

  const int niter = K / TILE_K;   // >= 2 for all our shapes
  int buf = 0;

  stage(0, 0);
  for (int it = 0; it < niter - 1; ++it) {
    stage(buf ^ 1, (it + 1) * TILE_K);
    asm volatile("s_wait_asynccnt 0x8" ::: "memory");   // this tile landed; next in flight
    __syncthreads();
    compute(buf);
    __syncthreads();               // all reads done before buffer is re-staged
    buf ^= 1;
  }
  asm volatile("s_wait_asynccnt 0x0" ::: "memory");
  __syncthreads();
  compute(buf);

  // epilogue: bias + activation; C/D layout: M = r + 8*half, N = l16
#pragma unroll
  for (int i = 0; i < 2; ++i) {
#pragma unroll
    for (int j = 0; j < 4; ++j) {
      const int col = n0 + waveN * 64 + j * 16 + l16;
      const float bv = bias[col];
#pragma unroll
      for (int r = 0; r < 8; ++r) {
        const int row = m0 + waveM * 32 + i * 16 + half * 8 + r;
        float val = acc[i][j][r] + bv;
        if (ACT == 1) val = val / (1.0f + __expf(-val));   // SiLU
        else if (ACT == 2) val = fmaxf(val, 0.0f);         // ReLU
        if (OUT_BF16) ((__bf16*)Cout)[(size_t)row * N + col] = (__bf16)val;
        else          ((float*)Cout)[(size_t)row * N + col] = val;
      }
    }
  }
}

// ---------------------------------------------------------------------------
// fp32 -> bf16 copy (n multiple of 4)
// ---------------------------------------------------------------------------
__global__ __launch_bounds__(256)
void cvt_bf16(const float* __restrict__ s, __bf16* __restrict__ d, int n)
{
  int i = (blockIdx.x * 256 + threadIdx.x) * 4;
  if (i < n) {
    const float4 v = *(const float4*)(s + i);
    d[i + 0] = (__bf16)v.x;  d[i + 1] = (__bf16)v.y;
    d[i + 2] = (__bf16)v.z;  d[i + 3] = (__bf16)v.w;
  }
}

// ---------------------------------------------------------------------------
// Wt[n][k] = (bf16)W[k][n]; W is [K,N], 32x32 LDS tiles; dims multiples of 32
// ---------------------------------------------------------------------------
__global__ __launch_bounds__(256)
void transpose_cvt(const float* __restrict__ W, __bf16* __restrict__ Wt, int K, int N)
{
  __shared__ float t[32][33];
  const int n0 = blockIdx.x * 32, k0 = blockIdx.y * 32;
  const int c = threadIdx.x & 31, r = threadIdx.x >> 5;   // 8 rows/pass
#pragma unroll
  for (int rr = 0; rr < 32; rr += 8)
    t[rr + r][c] = W[(size_t)(k0 + rr + r) * N + n0 + c];
  __syncthreads();
#pragma unroll
  for (int rr = 0; rr < 32; rr += 8)
    Wt[(size_t)(n0 + rr + r) * K + k0 + c] = (__bf16)t[c][rr + r];
}

// ---------------------------------------------------------------------------
// Per-row softmax entropy of h [B, H] -> ent[B]
// ---------------------------------------------------------------------------
__global__ __launch_bounds__(256)
void entropy_kernel(const float* __restrict__ h, float* __restrict__ ent, int H)
{
  __shared__ float s0[256], s1[256];
  const int tid = threadIdx.x;
  const float* row = h + (size_t)blockIdx.x * H;

  float mx = -3.4e38f;
  for (int i = tid; i < H; i += 256) mx = fmaxf(mx, row[i]);
  s0[tid] = mx; __syncthreads();
  for (int s = 128; s > 0; s >>= 1) {
    if (tid < s) s0[tid] = fmaxf(s0[tid], s0[tid + s]);
    __syncthreads();
  }
  mx = s0[0]; __syncthreads();

  float se = 0.f, she = 0.f;
  for (int i = tid; i < H; i += 256) {
    float v = row[i];
    float e = __expf(v - mx);
    se += e; she += v * e;
  }
  s0[tid] = se; s1[tid] = she; __syncthreads();
  for (int s = 128; s > 0; s >>= 1) {
    if (tid < s) { s0[tid] += s0[tid + s]; s1[tid] += s1[tid + s]; }
    __syncthreads();
  }
  if (tid == 0) ent[blockIdx.x] = (mx + logf(s0[0])) - s1[0] / s0[0];
}

// ---------------------------------------------------------------------------
// k_eff = min(floor(k * (1 + 0.5*mean(ent)/log(H))), H/2)
// ---------------------------------------------------------------------------
__global__ __launch_bounds__(256)
void keff_kernel(const float* __restrict__ ent, int nrows,
                 const int* __restrict__ kptr, int H, int* __restrict__ keff)
{
  __shared__ float s0[256];
  float sum = 0.f;
  for (int i = threadIdx.x; i < nrows; i += 256) sum += ent[i];
  s0[threadIdx.x] = sum; __syncthreads();
  for (int s = 128; s > 0; s >>= 1) {
    if (threadIdx.x < s) s0[threadIdx.x] += s0[threadIdx.x + s];
    __syncthreads();
  }
  if (threadIdx.x == 0) {
    float mean = s0[0] / (float)nrows;
    int k = *kptr;
    int ke = (int)floorf((float)k * (1.0f + 0.5f * mean / logf((float)H)));
    if (ke > H / 2) ke = H / 2;
    if (ke < 1) ke = 1;
    *keff = ke;
  }
}

// ---------------------------------------------------------------------------
// k-WTA: exact k-th largest per row via 4-pass radix select; mask in place
// (fp32 in d_out) and emit bf16 copy for the decoder GEMM.
// ---------------------------------------------------------------------------
__device__ __forceinline__ unsigned wta_key(float f) {
  unsigned s = __float_as_uint(f);
  unsigned u = (s & 0x80000000u) ? ~s : (s | 0x80000000u);  // ascending
  return ~u;                                                // descending
}
__device__ __forceinline__ float wta_unkey(unsigned key) {
  unsigned u = ~key;
  unsigned s = (u & 0x80000000u) ? (u & 0x7FFFFFFFu) : ~u;
  return __uint_as_float(s);
}

__global__ __launch_bounds__(256)
void wta_kernel(float* __restrict__ z, __bf16* __restrict__ zb,
                const int* __restrict__ keff, int H)
{
  __shared__ unsigned hist[256];
  __shared__ unsigned sh_prefix;
  __shared__ int      sh_rank;

  float*  row  = z  + (size_t)blockIdx.x * H;
  __bf16* rowb = zb + (size_t)blockIdx.x * H;
  const int tid = threadIdx.x;
  int rank = *keff - 1;
  unsigned prefix = 0;

  for (int shift = 24; shift >= 0; shift -= 8) {
    hist[tid] = 0;
    __syncthreads();
    const unsigned mask_hi = (shift == 24) ? 0u : (0xFFFFFFFFu << (shift + 8));
    for (int i = tid; i < H; i += 256) {
      unsigned key = wta_key(row[i]);
      if ((key & mask_hi) == prefix)
        atomicAdd(&hist[(key >> shift) & 255u], 1u);
    }
    __syncthreads();
    if (tid == 0) {
      unsigned cum = 0; int b = 0;
      for (; b < 255; ++b) {
        unsigned c = hist[b];
        if (cum + c > (unsigned)rank) break;
        cum += c;
      }
      sh_prefix = prefix | ((unsigned)b << shift);
      sh_rank   = rank - (int)cum;
    }
    __syncthreads();
    prefix = sh_prefix;
    rank   = sh_rank;
    __syncthreads();
  }

  const float thresh = wta_unkey(prefix);
  for (int i = tid; i < H; i += 256) {
    float v = row[i];
    float o = (v >= thresh) ? v : 0.0f;
    row[i]  = o;
    rowb[i] = (__bf16)o;
  }
}

// ---------------------------------------------------------------------------
// Orchestration. mode is a static Python 0 in the reference -> storage head.
// ---------------------------------------------------------------------------
extern "C" void kernel_launch(void* const* d_in, const int* in_sizes, int n_in,
                              void* d_out, int out_size, void* d_ws, size_t ws_size,
                              hipStream_t stream)
{
  const float* x         = (const float*)d_in[0];
  const float* W_storage = (const float*)d_in[1];
  const float* b_storage = (const float*)d_in[2];
  const float* W_enc     = (const float*)d_in[5];
  const float* b_enc     = (const float*)d_in[6];
  const float* W_dec     = (const float*)d_in[7];
  const float* b_dec     = (const float*)d_in[8];
  const int*   kptr      = (const int*)d_in[10];

  float* recon = (float*)d_out;                       // [B, IN]
  float* z     = (float*)d_out + (size_t)B_ * IN_;    // [B, HID]

  // workspace layout
  char* ws = (char*)d_ws;
  size_t o = 0;
  __bf16* x_bf   = (__bf16*)(ws + o); o += (size_t)B_  * SD_  * 2;   // 16 MB
  __bf16* WstT   = (__bf16*)(ws + o); o += (size_t)IN_ * SD_  * 2;   //  1 MB
  __bf16* WencT  = (__bf16*)(ws + o); o += (size_t)HID_* IN_  * 2;   //  4 MB
  __bf16* WdecT  = (__bf16*)(ws + o); o += (size_t)IN_ * HID_ * 2;   //  4 MB
  __bf16* xh_bf  = (__bf16*)(ws + o); o += (size_t)B_  * IN_  * 2;   //  8 MB
  __bf16* z_bf   = (__bf16*)(ws + o); o += (size_t)B_  * HID_ * 2;   // 64 MB
  float*  ent    = (float*) (ws + o); o += (size_t)B_ * 4;
  int*    keff   = (int*)   (ws + o);

  const dim3 blk(256);

  // pre-passes: bf16 conversions / weight transposes (K-major operands)
  cvt_bf16<<<(B_ * SD_) / (256 * 4), blk, 0, stream>>>(x, x_bf, B_ * SD_);
  transpose_cvt<<<dim3(IN_ / 32, SD_ / 32), blk, 0, stream>>>(W_storage, WstT, SD_, IN_);
  transpose_cvt<<<dim3(HID_ / 32, IN_ / 32), blk, 0, stream>>>(W_enc, WencT, IN_, HID_);
  transpose_cvt<<<dim3(IN_ / 32, HID_ / 32), blk, 0, stream>>>(W_dec, WdecT, HID_, IN_);

  // xh = silu(x @ W_storage + b_storage)  -> bf16
  gemm_async<1, true><<<dim3(IN_ / TILE_N, B_ / TILE_M), blk, 0, stream>>>(
      x_bf, WstT, b_storage, xh_bf, B_, IN_, SD_);
  // h = relu(xh @ W_enc + b_enc)          -> fp32 into z region of d_out
  gemm_async<2, false><<<dim3(HID_ / TILE_N, B_ / TILE_M), blk, 0, stream>>>(
      xh_bf, WencT, b_enc, z, B_, HID_, IN_);
  // entropy -> k_eff -> k-WTA (in-place fp32 + bf16 copy)
  entropy_kernel<<<B_, blk, 0, stream>>>(z, ent, HID_);
  keff_kernel<<<1, blk, 0, stream>>>(ent, B_, kptr, HID_, keff);
  wta_kernel<<<B_, blk, 0, stream>>>(z, z_bf, keff, HID_);
  // recon = z_sparse @ W_dec + b_dec      -> fp32
  gemm_async<0, false><<<dim3(IN_ / TILE_N, B_ / TILE_M), blk, 0, stream>>>(
      z_bf, WdecT, b_dec, recon, B_, IN_, HID_);
}
